// HeteroGNN_61770219651569
// MI455X (gfx1250) — matrix operand order repaired and checked by reading, not compiled, not run on hardware
//
#include <hip/hip_runtime.h>

typedef __attribute__((ext_vector_type(16))) _Float16 v16h;
typedef __attribute__((ext_vector_type(8)))  float    v8f;

// ---------------------------------------------------------------------------
// WMMA helpers (wave32, layouts per CDNA5 ISA 7.12.2)
// ---------------------------------------------------------------------------

static __device__ __forceinline__ v8f wmma_f16(v16h a, v16h b, v8f c) {
  // D = A x B + C, 16x16x32, f32 accumulate
  return __builtin_amdgcn_wmma_f32_16x16x32_f16(false, a, false, b, (short)0, c, false, false);
}

// A fragment: 16x32 (MxK) f16 from row-major f32.
// Lanes 0-15: row=lane, K = k0+{0..7,16..23}; lanes 16-31: row=lane-16, K = k0+{8..15,24..31}.
static __device__ __forceinline__ v16h load_a_frag_f32(const float* __restrict__ Xrow,
                                                       int k0, int hi, float scale) {
  const float* p = Xrow + k0 + hi * 8;
  float4 a0 = *(const float4*)(p + 0);
  float4 a1 = *(const float4*)(p + 4);
  float4 a2 = *(const float4*)(p + 16);
  float4 a3 = *(const float4*)(p + 20);
  v16h a;
  a[0]  = (_Float16)(a0.x * scale); a[1]  = (_Float16)(a0.y * scale);
  a[2]  = (_Float16)(a0.z * scale); a[3]  = (_Float16)(a0.w * scale);
  a[4]  = (_Float16)(a1.x * scale); a[5]  = (_Float16)(a1.y * scale);
  a[6]  = (_Float16)(a1.z * scale); a[7]  = (_Float16)(a1.w * scale);
  a[8]  = (_Float16)(a2.x * scale); a[9]  = (_Float16)(a2.y * scale);
  a[10] = (_Float16)(a2.z * scale); a[11] = (_Float16)(a2.w * scale);
  a[12] = (_Float16)(a3.x * scale); a[13] = (_Float16)(a3.y * scale);
  a[14] = (_Float16)(a3.z * scale); a[15] = (_Float16)(a3.w * scale);
  return a;
}

// B fragment from pre-packed fragment-major f16 weights (global or LDS pointer):
// 32 B contiguous per lane -> two 128-bit loads, no converts in the hot loop.
static __device__ __forceinline__ v16h load_b_frag_packed(const _Float16* __restrict__ P,
                                                          int frag_idx, int lane) {
  const uint4* p = (const uint4*)(P + ((long long)frag_idx * 32 + lane) * 16);
  union { uint4 u[2]; v16h h; } f;
  f.u[0] = p[0];
  f.u[1] = p[1];
  return f.h;
}

// ---------------------------------------------------------------------------
// Weight packing: row-major f32 W[K x N] -> f16 fragments [kk][t][lane][i]
//   krow = kk*32 + (lane>>4)*8 + (i&7) + (i>>3)*16 ; col = t*16 + (lane&15)
// Matches the WMMA 32x16 f16 B-matrix VGPR layout.
// ---------------------------------------------------------------------------
__global__ __launch_bounds__(256) void pack_w_kernel(const float* __restrict__ W,
                                                     _Float16* __restrict__ out,
                                                     int K, int N) {
  const int idx = blockIdx.x * blockDim.x + threadIdx.x;
  if (idx >= K * N) return;
  const int i    = idx & 15;
  const int lane = (idx >> 4) & 31;
  const int rest = idx >> 9;
  const int nt   = N >> 4;
  const int t    = rest % nt;
  const int kk   = rest / nt;
  const int krow = kk * 32 + (lane >> 4) * 8 + (i & 7) + (i >> 3) * 16;
  const int col  = t * 16 + (lane & 15);
  out[idx] = (_Float16)W[(long long)krow * N + col];
}

// ---------------------------------------------------------------------------
// Node embedding  out[n x 64] = relu(X[n x K] @ W + b), packed f16 W.
// One wave per 16-row tile. Ping-pong 2-stage pipeline (unroll-by-2): buffer
// sets alternate roles structurally, so no inter-stage register copies; next
// stage's loads always issue before the current stage's WMMAs.
// ---------------------------------------------------------------------------
__global__ __launch_bounds__(256) void embed_kernel(const float* __restrict__ X,
                                                    const _Float16* __restrict__ Wp,
                                                    const float* __restrict__ bias,
                                                    float* __restrict__ out,
                                                    int nrows, int K) {
  const int lane = threadIdx.x & 31;
  const int wave = threadIdx.x >> 5;
  const int tile = blockIdx.x * 8 + wave;
  const int row0 = tile * 16;
  if (row0 >= nrows) return;                 // wave-uniform: EXEC all-1s for WMMA
  const int hi = lane >> 4;
  const int m  = lane & 15;
  int arow = row0 + m; if (arow >= nrows) arow = nrows - 1;
  const float* Xrow = X + (long long)arow * K;

  v8f acc[4] = {};
  // prologue: stage k = 0 into buffer set 0
  v16h a0 = load_a_frag_f32(Xrow, 0, hi, 1.0f);
  v16h b0[4];
#pragma unroll
  for (int t = 0; t < 4; ++t) b0[t] = load_b_frag_packed(Wp, t, lane);

  for (int k0 = 0; k0 < K; k0 += 64) {
    __builtin_prefetch(Xrow + k0 + 256, 0, 1);       // global_prefetch_b8, A stream
    // load stage k0+32 into set 1 (clamped dummy on tail)
    const int k1  = k0 + 32;
    const int k1c = (k1 < K) ? k1 : k0;
    v16h a1 = load_a_frag_f32(Xrow, k1c, hi, 1.0f);
    v16h b1[4];
    const int fb1 = (k1c >> 5) * 4;
#pragma unroll
    for (int t = 0; t < 4; ++t) b1[t] = load_b_frag_packed(Wp, fb1 + t, lane);
    // compute stage k0 (operands resident)
#pragma unroll
    for (int t = 0; t < 4; ++t) acc[t] = wmma_f16(a0, b0[t], acc[t]);
    // load stage k0+64 into set 0 (clamped dummy on tail)
    const int k2  = k0 + 64;
    const int k2c = (k2 < K) ? k2 : k1c;
    a0 = load_a_frag_f32(Xrow, k2c, hi, 1.0f);
    const int fb2 = (k2c >> 5) * 4;
#pragma unroll
    for (int t = 0; t < 4; ++t) b0[t] = load_b_frag_packed(Wp, fb2 + t, lane);
    // compute stage k0+32 (wave-uniform guard; EXEC untouched)
    if (k1 < K) {
#pragma unroll
      for (int t = 0; t < 4; ++t) acc[t] = wmma_f16(a1, b1[t], acc[t]);
    }
  }

  const int mbase = row0 + hi * 8;
#pragma unroll
  for (int t = 0; t < 4; ++t) {
    const int n = t * 16 + m;
    const float bv = bias[n];
#pragma unroll
    for (int r = 0; r < 8; ++r) {
      const int M = mbase + r;
      if (M < nrows) {
        float v = acc[t][r] + bv;
        out[(long long)M * 64 + n] = v > 0.0f ? v : 0.0f;
      }
    }
  }
}

// ---------------------------------------------------------------------------
// In-degree counts -> 1/max(cnt,1)
// ---------------------------------------------------------------------------
__global__ __launch_bounds__(256) void count_kernel(const int* __restrict__ dst, int E,
                                                    float* __restrict__ cnt) {
  int e = blockIdx.x * blockDim.x + threadIdx.x;
  if (e < E) atomicAdd(cnt + dst[e], 1.0f);
}

__global__ __launch_bounds__(256) void inv_kernel(float* __restrict__ cnt, int n) {
  int i = blockIdx.x * blockDim.x + threadIdx.x;
  if (i < n) {
    float c = cnt[i];
    cnt[i] = 1.0f / (c > 1.0f ? c : 1.0f);
  }
}

// ---------------------------------------------------------------------------
// Edge scatter-add  acc[dst[e]] += xsrc[src[e]]  (H = 64); L2-resident tables.
// ---------------------------------------------------------------------------
__global__ __launch_bounds__(256) void scatter_kernel(const float* __restrict__ xsrc,
                                                      const int* __restrict__ src,
                                                      const int* __restrict__ dst,
                                                      int E, float* __restrict__ acc) {
  long long idx = (long long)blockIdx.x * blockDim.x + threadIdx.x;
  if (idx >= (long long)E * 16) return;
  const int e = (int)(idx >> 4);
  const int c = ((int)idx & 15) * 4;
  const int s = src[e], d = dst[e];
  const float4 v = *(const float4*)(xsrc + (long long)s * 64 + c);
  float* p = acc + (long long)d * 64 + c;
  atomicAdd(p + 0, v.x);
  atomicAdd(p + 1, v.y);
  atomicAdd(p + 2, v.z);
  atomicAdd(p + 3, v.w);
}

// ---------------------------------------------------------------------------
// Fused SAGE update (in place):  x = relu((agg*inv) @ Wl + bl + x @ Wr) + x
// Packed f16 Wl/Wr staged into LDS via CDNA5 async global->LDS copies
// (global_load_async_to_lds_b128 + s_wait_asynccnt), then read as ds_load_b128.
// Staging runs on ALL threads before the tile-bounds return (no barrier hazard).
// ---------------------------------------------------------------------------
__global__ __launch_bounds__(256) void sage_update_kernel(const float* __restrict__ agg,
                                                          const float* __restrict__ inv,
                                                          float* __restrict__ x,
                                                          const _Float16* __restrict__ Wlp,
                                                          const float* __restrict__ bl,
                                                          const _Float16* __restrict__ Wrp,
                                                          int nrows) {
  __shared__ __align__(16) _Float16 sWl[64 * 64];   // 8 KB
  __shared__ __align__(16) _Float16 sWr[64 * 64];   // 8 KB

  const int tid = threadIdx.x;
  // Cooperative async stage: 2 x 8KB = 1024 x 16B chunks over 256 threads.
  // LDS byte offset = low 32 bits of the flat shared-space pointer (ISA 10.2).
#pragma unroll
  for (int i = 0; i < 2; ++i) {
    const int chunk = tid + i * 256;                 // 0..511 per matrix
    unsigned int       l0 = (unsigned int)(unsigned long long)(sWl + chunk * 8);
    unsigned long long g0 = (unsigned long long)(Wlp + chunk * 8);
    asm volatile("global_load_async_to_lds_b128 %0, %1, off" :: "v"(l0), "v"(g0) : "memory");
    unsigned int       l1 = (unsigned int)(unsigned long long)(sWr + chunk * 8);
    unsigned long long g1 = (unsigned long long)(Wrp + chunk * 8);
    asm volatile("global_load_async_to_lds_b128 %0, %1, off" :: "v"(l1), "v"(g1) : "memory");
  }
  asm volatile("s_wait_asynccnt 0" ::: "memory");
  __syncthreads();

  const int lane = tid & 31;
  const int wave = tid >> 5;
  const int tile = blockIdx.x * 8 + wave;
  const int row0 = tile * 16;
  if (row0 >= nrows) return;                 // wave-uniform (after the barrier)
  const int hi = lane >> 4;
  const int m  = lane & 15;
  int arow = row0 + m; if (arow >= nrows) arow = nrows - 1;
  const float sc = inv[arow];
  const float* Arow = agg + (long long)arow * 64;
  const float* Xrow = x   + (long long)arow * 64;

  v8f acc[4] = {};
#pragma unroll
  for (int k0 = 0; k0 < 64; k0 += 32) {
    v16h am = load_a_frag_f32(Arow, k0, hi, sc);     // mean aggregate
    v16h ax = load_a_frag_f32(Xrow, k0, hi, 1.0f);   // self features
    const int fb = (k0 >> 5) * 4;
#pragma unroll
    for (int t = 0; t < 4; ++t) {
      acc[t] = wmma_f16(am, load_b_frag_packed(sWl, fb + t, lane), acc[t]);
      acc[t] = wmma_f16(ax, load_b_frag_packed(sWr, fb + t, lane), acc[t]);
    }
  }
  const int mbase = row0 + hi * 8;
#pragma unroll
  for (int t = 0; t < 4; ++t) {
    const int n = t * 16 + m;
    const float bv = bl[n];
#pragma unroll
    for (int r = 0; r < 8; ++r) {
      const int M = mbase + r;
      if (M < nrows) {
        float v = acc[t][r] + bv;
        v = v > 0.0f ? v : 0.0f;
        float* px = x + (long long)M * 64 + n;
        *px = v + *px;                        // residual; row owned by this tile only
      }
    }
  }
}

// ---------------------------------------------------------------------------
// Fused head  out = relu(x @ W1[64x32] + b1) @ W2[32x1] + b2
// 4 WMMAs per tile + 16-lane shfl_xor reduction for the [32x1] dot.
// ---------------------------------------------------------------------------
__global__ __launch_bounds__(256) void head_kernel(const float* __restrict__ x,
                                                   const _Float16* __restrict__ W1p,
                                                   const float* __restrict__ b1,
                                                   const float* __restrict__ W2,
                                                   const float* __restrict__ b2,
                                                   float* __restrict__ out, int nrows) {
  const int lane = threadIdx.x & 31;
  const int wave = threadIdx.x >> 5;
  const int tile = blockIdx.x * 8 + wave;
  const int row0 = tile * 16;
  if (row0 >= nrows) return;
  const int hi = lane >> 4;
  const int m  = lane & 15;
  int arow = row0 + m; if (arow >= nrows) arow = nrows - 1;
  const float* Xrow = x + (long long)arow * 64;

  v8f acc[2] = {};
#pragma unroll
  for (int k0 = 0; k0 < 64; k0 += 32) {
    v16h a = load_a_frag_f32(Xrow, k0, hi, 1.0f);
    const int fb = (k0 >> 5) * 2;
    acc[0] = wmma_f16(a, load_b_frag_packed(W1p, fb + 0, lane), acc[0]);
    acc[1] = wmma_f16(a, load_b_frag_packed(W1p, fb + 1, lane), acc[1]);
  }
  const float b1a = b1[m], b1b = b1[16 + m];
  const float w2a = W2[m], w2b = W2[16 + m];
  float p[8];
#pragma unroll
  for (int r = 0; r < 8; ++r) {
    float h0 = acc[0][r] + b1a; h0 = h0 > 0.0f ? h0 : 0.0f;
    float h1 = acc[1][r] + b1b; h1 = h1 > 0.0f ? h1 : 0.0f;
    p[r] = h0 * w2a + h1 * w2b;
  }
#pragma unroll
  for (int r = 0; r < 8; ++r) {    // reduce across the 16 lanes of each half-wave
    p[r] += __shfl_xor(p[r], 1, 32);
    p[r] += __shfl_xor(p[r], 2, 32);
    p[r] += __shfl_xor(p[r], 4, 32);
    p[r] += __shfl_xor(p[r], 8, 32);
  }
  if (m == 0) {
    const float bo = b2[0];
    const int mbase = row0 + hi * 8;
#pragma unroll
    for (int r = 0; r < 8; ++r) {
      const int M = mbase + r;
      if (M < nrows) out[M] = p[r] + bo;
    }
  }
}

// ---------------------------------------------------------------------------
// Host orchestration (graph-capture safe: async launches + memsetAsync only)
// ---------------------------------------------------------------------------
extern "C" void kernel_launch(void* const* d_in, const int* in_sizes, int n_in,
                              void* d_out, int out_size, void* d_ws, size_t ws_size,
                              hipStream_t stream) {
  const float* x_comp = (const float*)d_in[0];
  const float* x_tgt  = (const float*)d_in[1];
  const int*   src_ct = (const int*)d_in[2];
  const int*   dst_ct = (const int*)d_in[3];
  const int*   src_tc = (const int*)d_in[4];
  const int*   dst_tc = (const int*)d_in[5];
  const float* Wc     = (const float*)d_in[6];
  const float* bc     = (const float*)d_in[7];
  const float* Wt     = (const float*)d_in[8];
  const float* bt     = (const float*)d_in[9];
  const float* Wl_ct  = (const float*)d_in[10];
  const float* bl_ct  = (const float*)d_in[11];
  const float* Wr_ct  = (const float*)d_in[12];
  const float* Wl_tc  = (const float*)d_in[13];
  const float* bl_tc  = (const float*)d_in[14];
  const float* Wr_tc  = (const float*)d_in[15];
  const float* Wo1    = (const float*)d_in[16];
  const float* bo1    = (const float*)d_in[17];
  const float* Wo2    = (const float*)d_in[18];
  const float* bo2    = (const float*)d_in[19];

  const int H   = in_sizes[7];            // 64
  const int D_C = in_sizes[6] / H;        // 1024
  const int D_T = in_sizes[8] / H;        // 1280
  const int N_C = in_sizes[0] / D_C;      // 100000
  const int N_T = in_sizes[1] / D_T;      // 5000
  const int E   = in_sizes[2];            // 2000000
  const int L   = in_sizes[10] / (H * H); // 3
  const int H2  = in_sizes[16] / H;       // 32

  // workspace partition (256B aligned chunks)
  char* ws = (char*)d_ws;
  size_t off = 0;
  auto alloc = [&](size_t bytes) -> char* {
    char* p = ws + off;
    off += (bytes + 255) & ~(size_t)255;
    return p;
  };
  float* xc    = (float*)alloc((size_t)N_C * H * sizeof(float));
  float* xt    = (float*)alloc((size_t)N_T * H * sizeof(float));
  float* agg_c = (float*)alloc((size_t)N_C * H * sizeof(float));
  float* agg_t = (float*)alloc((size_t)N_T * H * sizeof(float));
  float* inv_c = (float*)alloc((size_t)N_C * sizeof(float));
  float* inv_t = (float*)alloc((size_t)N_T * sizeof(float));
  _Float16* pWc    = (_Float16*)alloc((size_t)D_C * H * 2);
  _Float16* pWt    = (_Float16*)alloc((size_t)D_T * H * 2);
  _Float16* pWl_ct = (_Float16*)alloc((size_t)L * H * H * 2);
  _Float16* pWr_ct = (_Float16*)alloc((size_t)L * H * H * 2);
  _Float16* pWl_tc = (_Float16*)alloc((size_t)L * H * H * 2);
  _Float16* pWr_tc = (_Float16*)alloc((size_t)L * H * H * 2);
  _Float16* pWo1   = (_Float16*)alloc((size_t)H * H2 * 2);
  (void)ws_size; (void)n_in; (void)out_size;

  const int T = 256;
  auto cdiv    = [](int a, int b) { return (a + b - 1) / b; };
  auto tblocks = [](int n) { return (((n + 15) / 16) + 7) / 8; };  // 16-row tiles, 8 waves/block

  // 0) pack weights to f16 fragment-major once per call (~400 KB total)
  pack_w_kernel<<<cdiv(D_C * H, T), T, 0, stream>>>(Wc, pWc, D_C, H);
  pack_w_kernel<<<cdiv(D_T * H, T), T, 0, stream>>>(Wt, pWt, D_T, H);
  for (int l = 0; l < L; ++l) {   // per-layer chunks are self-contained (H multiple of 32)
    pack_w_kernel<<<cdiv(H * H, T), T, 0, stream>>>(Wl_ct + (size_t)l * H * H, pWl_ct + (size_t)l * H * H, H, H);
    pack_w_kernel<<<cdiv(H * H, T), T, 0, stream>>>(Wr_ct + (size_t)l * H * H, pWr_ct + (size_t)l * H * H, H, H);
    pack_w_kernel<<<cdiv(H * H, T), T, 0, stream>>>(Wl_tc + (size_t)l * H * H, pWl_tc + (size_t)l * H * H, H, H);
    pack_w_kernel<<<cdiv(H * H, T), T, 0, stream>>>(Wr_tc + (size_t)l * H * H, pWr_tc + (size_t)l * H * H, H, H);
  }
  pack_w_kernel<<<cdiv(H * H2, T), T, 0, stream>>>(Wo1, pWo1, H, H2);

  // 1) in-degree -> 1/max(cnt,1)  (layer-invariant)
  hipMemsetAsync(inv_c, 0, (size_t)N_C * sizeof(float), stream);
  hipMemsetAsync(inv_t, 0, (size_t)N_T * sizeof(float), stream);
  count_kernel<<<cdiv(E, T), T, 0, stream>>>(dst_ct, E, inv_t);
  count_kernel<<<cdiv(E, T), T, 0, stream>>>(dst_tc, E, inv_c);
  inv_kernel<<<cdiv(N_T, T), T, 0, stream>>>(inv_t, N_T);
  inv_kernel<<<cdiv(N_C, T), T, 0, stream>>>(inv_c, N_C);

  // 2) input embeddings (big HBM-streaming WMMA GEMMs, ping-pong pipelined)
  embed_kernel<<<tblocks(N_C), T, 0, stream>>>(x_comp, pWc, bc, xc, N_C, D_C);
  embed_kernel<<<tblocks(N_T), T, 0, stream>>>(x_tgt,  pWt, bt, xt, N_T, D_T);

  const long long sc_total = (long long)E * 16;
  const int sc_blocks = (int)((sc_total + T - 1) / T);

  // 3) message-passing layers: both scatters read OLD states, then in-place updates
  for (int l = 0; l < L; ++l) {
    hipMemsetAsync(agg_t, 0, (size_t)N_T * H * sizeof(float), stream);
    hipMemsetAsync(agg_c, 0, (size_t)N_C * H * sizeof(float), stream);
    scatter_kernel<<<sc_blocks, T, 0, stream>>>(xc, src_ct, dst_ct, E, agg_t);
    scatter_kernel<<<sc_blocks, T, 0, stream>>>(xt, src_tc, dst_tc, E, agg_c);
    sage_update_kernel<<<tblocks(N_T), T, 0, stream>>>(
        agg_t, inv_t, xt, pWl_ct + (size_t)l * H * H, bl_ct + (size_t)l * H,
        pWr_ct + (size_t)l * H * H, N_T);
    sage_update_kernel<<<tblocks(N_C), T, 0, stream>>>(
        agg_c, inv_c, xc, pWl_tc + (size_t)l * H * H, bl_tc + (size_t)l * H,
        pWr_tc + (size_t)l * H * H, N_C);
  }

  // 4) fused output head
  head_kernel<<<tblocks(N_C), T, 0, stream>>>(xc, pWo1, bo1, Wo2, bo2, (float*)d_out, N_C);
}
